// GraphAttention_7499012899176
// MI455X (gfx1250) — matrix operand order repaired and checked by reading, not compiled, not run on hardware
//
#include <hip/hip_runtime.h>

typedef __attribute__((ext_vector_type(16))) _Float16 v16h;
typedef __attribute__((ext_vector_type(8)))  float    v8f;
typedef __attribute__((ext_vector_type(2)))  _Float16 h2;
typedef __attribute__((ext_vector_type(4)))  int      v4i;

#define NN 8192
#define DIN 256
#define DOUT 64
#define NEG_INF (-9.0e15f)
#define SLOPE 0.2f

// workspace layout (bytes)
#define Z_OFF   0u                            // f32 z[8192][64]            (2 MB)
#define ZHT_OFF (Z_OFF + NN*DOUT*4u)          // f16 z_ht[64][8192] (T)     (1 MB)
#define SI_OFF  (ZHT_OFF + NN*DOUT*2u)        // f32 si[8192]
#define SJ_OFF  (SI_OFF + NN*4u)              // f32 sj[8192]
#define WHT_OFF (SJ_OFF + NN*4u)              // f16 w_ht[64][256] (T)      (32 KB)

// ---- gfx1250 async global->LDS probe ---------------------------------------
#if defined(__has_builtin)
#  if __has_builtin(__builtin_amdgcn_global_load_async_to_lds_b128)
#    define HAVE_ASYNC 1
#  endif
#endif
#ifndef HAVE_ASYNC
#  define HAVE_ASYNC 0
#endif

#if HAVE_ASYNC
typedef __attribute__((address_space(1))) v4i gv4i;   // global int4
typedef __attribute__((address_space(3))) v4i lv4i;   // LDS int4
#endif

#if defined(__has_builtin)
#  if __has_builtin(__builtin_amdgcn_s_wait_asynccnt)
#    define WAIT_ASYNC(n) do { __builtin_amdgcn_s_wait_asynccnt(n); asm volatile("" ::: "memory"); } while (0)
#  endif
#endif
#ifndef WAIT_ASYNC
#  define WAIT_ASYNC(n) asm volatile("s_wait_asynccnt %0" :: "n"(n) : "memory")
#endif

#if defined(__has_builtin)
#  if __has_builtin(__builtin_amdgcn_s_wait_dscnt)
#    define WAIT_DS0() do { __builtin_amdgcn_s_wait_dscnt(0); asm volatile("" ::: "memory"); } while (0)
#  endif
#endif
#ifndef WAIT_DS0
#  define WAIT_DS0() asm volatile("s_wait_dscnt 0" ::: "memory")
#endif

// ---------------------------------------------------------------- prep: W -> f16 transposed
__global__ void prep_w(const float* __restrict__ W, _Float16* __restrict__ w_ht) {
    int t = blockIdx.x * blockDim.x + threadIdx.x;   // 0..16383
    int n = t >> 8;          // output col 0..63
    int k = t & 255;         // input dim 0..255
    w_ht[n * 256 + k] = (_Float16)W[k * 64 + n];
}

// ---------------------------------------------------------------- z = x @ W  (one wave per 16-row tile)
__global__ __launch_bounds__(32) void gemm_z(const float* __restrict__ x,
                                             const _Float16* __restrict__ w_ht,
                                             float* __restrict__ z,
                                             _Float16* __restrict__ z_ht) {
    const int L = threadIdx.x;
    const int rowBase = blockIdx.x * 16;
    const int M = L & 15;
    const int khalf = (L >> 4) * 8;      // 0 for lanes 0-15, 8 for lanes 16-31

    v8f acc[4] = {};

    for (int k0 = 0; k0 < DIN; k0 += 32) {
        // A fragment: x rows, 16-bit A-matrix 16x32 layout
        v16h af;
        const float* xr = x + (size_t)(rowBase + M) * DIN + k0;
#pragma unroll
        for (int i = 0; i < 8; ++i) {
            int kk = ((i & 4) << 2) + khalf + 2 * (i & 3);
            float2 xv = *(const float2*)(xr + kk);
            af[2 * i]     = (_Float16)xv.x;
            af[2 * i + 1] = (_Float16)xv.y;
        }
#pragma unroll
        for (int n = 0; n < 4; ++n) {
            // B fragment: W columns (transposed storage -> contiguous pairs)
            v16h bf;
            const _Float16* wp = w_ht + (size_t)(n * 16 + M) * 256 + k0;
#pragma unroll
            for (int i = 0; i < 8; ++i) {
                h2 wv = *(const h2*)(wp + ((i & 4) << 2) + khalf + 2 * (i & 3));
                bf[2 * i]     = wv.x;
                bf[2 * i + 1] = wv.y;
            }
            acc[n] = __builtin_amdgcn_wmma_f32_16x16x32_f16(
                false, af, false, bf, (short)0, acc[n], false, false);
        }
    }

    // C layout: element r -> row r + khalf, col n*16 + (L&15)
#pragma unroll
    for (int n = 0; n < 4; ++n)
#pragma unroll
        for (int r = 0; r < 8; ++r) {
            int row = rowBase + khalf + r;
            int col = n * 16 + M;
            float v = acc[n][r];
            z[(size_t)row * DOUT + col] = v;
            z_ht[(size_t)col * NN + row] = (_Float16)v;
        }
}

// ---------------------------------------------------------------- si = z@a1, sj = z@a2
__global__ void scores(const float* __restrict__ z, const float* __restrict__ a,
                       float* __restrict__ si, float* __restrict__ sj) {
    int row = blockIdx.x * blockDim.x + threadIdx.x;
    if (row >= NN) return;
    float s1 = 0.f, s2 = 0.f;
#pragma unroll 8
    for (int c = 0; c < DOUT; ++c) {
        float v = z[(size_t)row * DOUT + c];
        s1 += v * a[c];
        s2 += v * a[DOUT + c];
    }
    si[row] = s1;
    sj[row] = s2;
}

// ---------------------------------------------------------------- fused masked-softmax + alpha@z
// Block: 4 waves, 16 output rows. Each wave owns a 2048-column chunk and keeps
// online-softmax state (m, l, acc). adj tiles (16x64 ints) are double-buffered
// through LDS with async global->LDS loads (ASYNCcnt), rows padded to 68 dwords
// to spread the 16-row column reads over 16 LDS banks.
#define TILE_C   64            // columns per staged tile
#define ROW_PAD  68            // dwords per padded LDS row
#define TILE_DW  (16 * ROW_PAD)

__global__ __launch_bounds__(128) void gat_fused(const int* __restrict__ adj,
                                                 const float* __restrict__ si,
                                                 const float* __restrict__ sj,
                                                 const _Float16* __restrict__ z_ht,
                                                 const float* __restrict__ b,
                                                 float* __restrict__ out) {
    __shared__ float accS[4][16][64];
    __shared__ float mS[4][16];
    __shared__ float lS[4][16];
#if HAVE_ASYNC
    __shared__ int adjS[4 * 2 * TILE_DW];    // [wave][buf][16 x 68]
#endif

    const int L = threadIdx.x & 31;
    const int w = threadIdx.x >> 5;
    const int rowBase = blockIdx.x * 16;
    const int M = L & 15;
    const int khalf = (L >> 4) * 8;

    const float si_r = si[rowBase + M];
    const int* adjRow  = adj + (size_t)(rowBase + M) * NN;   // fallback path
    const int* adjTile = adj + (size_t)rowBase * NN;         // async path
    (void)adjRow; (void)adjTile;

    float m = NEG_INF, l = 0.f;
    v8f acc[4] = {};

    const int jBeg = w * (NN / 4);
    const int NT = (NN / 4) / TILE_C;                        // 32 tiles per wave

#if HAVE_ASYNC
    // issue one 16x64 tile (4 KB) as 8 coalesced 16B/lane async transfers
#define ISSUE_TILE(tt, bb)                                                         \
    do {                                                                           \
        const int _base = (w * 2 + (bb)) * TILE_DW;                                \
        _Pragma("unroll")                                                          \
        for (int q = 0; q < 8; ++q) {                                              \
            int idx = q * 32 + L;                                                  \
            int row = idx >> 4, chunk = idx & 15;                                  \
            const int* g = adjTile + (size_t)row * NN + jBeg + (tt) * TILE_C +     \
                           chunk * 4;                                              \
            int* lp = &adjS[_base + row * ROW_PAD + chunk * 4];                    \
            __builtin_amdgcn_global_load_async_to_lds_b128((gv4i*)g, (lv4i*)lp,    \
                                                           0, 0);                  \
        }                                                                          \
    } while (0)

    ISSUE_TILE(0, 0);
    ISSUE_TILE(1, 1);
#endif

    for (int t = 0; t < NT; ++t) {
        const int j0 = jBeg + t * TILE_C;
#if HAVE_ASYNC
        if (t < NT - 1) WAIT_ASYNC(8); else WAIT_ASYNC(0);
        const int bufBase = (w * 2 + (t & 1)) * TILE_DW;
#endif
#pragma unroll
        for (int s = 0; s < 2; ++s) {
            const int jb = j0 + s * 32;
            // masked leaky-relu scores for this lane's 16 (row, col) slots
            float ev[16];
            float tmax = NEG_INF;
#pragma unroll
            for (int i = 0; i < 8; ++i) {
                int kk = ((i & 4) << 2) + khalf + 2 * (i & 3);
#if HAVE_ASYNC
                int2 ad = *(const int2*)&adjS[bufBase + M * ROW_PAD + s * 32 + kk];
#else
                int2 ad = *(const int2*)(adjRow + jb + kk);
#endif
                float2 sv = *(const float2*)(sj + jb + kk);
                float e0 = si_r + sv.x; e0 = e0 > 0.f ? e0 : SLOPE * e0;
                float e1 = si_r + sv.y; e1 = e1 > 0.f ? e1 : SLOPE * e1;
                e0 = ad.x > 0 ? e0 : NEG_INF;
                e1 = ad.y > 0 ? e1 : NEG_INF;
                ev[2 * i]     = e0;
                ev[2 * i + 1] = e1;
                tmax = fmaxf(tmax, fmaxf(e0, e1));
            }
            // per-row tile max: partner lane holds other half of K
            tmax = fmaxf(tmax, __shfl_xor(tmax, 16, 32));
            float m_new = fmaxf(m, tmax);
            float scale = __expf(m - m_new);

            // broadcast rescale factors to C-layout rows (lane r holds row r's stats)
            float fr[8];
#pragma unroll
            for (int r = 0; r < 8; ++r) fr[r] = __shfl(scale, r + khalf, 32);
#pragma unroll
            for (int n = 0; n < 4; ++n)
#pragma unroll
                for (int r = 0; r < 8; ++r) acc[n][r] *= fr[r];

            // p values -> A fragment (f16), row-sum for l
            float psum = 0.f;
            v16h af;
#pragma unroll
            for (int idx = 0; idx < 16; ++idx) {
                float p = __expf(ev[idx] - m_new);
                psum += p;
                af[idx] = (_Float16)p;
            }
            l = l * scale + psum + __shfl_xor(psum, 16, 32);
            m = m_new;

            // acc += p @ z_tile  (4 N-tiles of 16)
#pragma unroll
            for (int n = 0; n < 4; ++n) {
                v16h bf;
                const _Float16* zp = z_ht + (size_t)(n * 16 + M) * NN + jb;
#pragma unroll
                for (int i = 0; i < 8; ++i) {
                    h2 zv = *(const h2*)(zp + ((i & 4) << 2) + khalf + 2 * (i & 3));
                    bf[2 * i]     = zv.x;
                    bf[2 * i + 1] = zv.y;
                }
                acc[n] = __builtin_amdgcn_wmma_f32_16x16x32_f16(
                    false, af, false, bf, (short)0, acc[n], false, false);
            }
        }
#if HAVE_ASYNC
        if (t + 2 < NT) {
            WAIT_DS0();                 // our LDS reads of this buffer are done
            ISSUE_TILE(t + 2, t & 1);
        }
#endif
    }

    // dump wave-local state to LDS
    if (L < 16) { mS[w][L] = m; lS[w][L] = l; }
#pragma unroll
    for (int n = 0; n < 4; ++n)
#pragma unroll
        for (int r = 0; r < 8; ++r)
            accS[w][khalf + r][n * 16 + M] = acc[n][r];
    __syncthreads();

    // cross-wave merge: rescale to global row max, normalize, bias, relu
    for (int t = threadIdx.x; t < 16 * 64; t += 128) {
        int row = t >> 6, col = t & 63;
        float m0 = mS[0][row], m1 = mS[1][row], m2 = mS[2][row], m3 = mS[3][row];
        float ms = fmaxf(fmaxf(m0, m1), fmaxf(m2, m3));
        float f0 = __expf(m0 - ms), f1 = __expf(m1 - ms);
        float f2 = __expf(m2 - ms), f3 = __expf(m3 - ms);
        float s = accS[0][row][col] * f0 + accS[1][row][col] * f1 +
                  accS[2][row][col] * f2 + accS[3][row][col] * f3;
        float den = lS[0][row] * f0 + lS[1][row] * f1 + lS[2][row] * f2 + lS[3][row] * f3;
        float o = s / den + b[col];
        out[(size_t)(rowBase + row) * DOUT + col] = o > 0.f ? o : 0.f;
    }
}

// ---------------------------------------------------------------- launcher
extern "C" void kernel_launch(void* const* d_in, const int* in_sizes, int n_in,
                              void* d_out, int out_size, void* d_ws, size_t ws_size,
                              hipStream_t stream) {
    const float* x  = (const float*)d_in[0];
    const int*   adj= (const int*)  d_in[1];
    const float* W  = (const float*)d_in[2];
    const float* a  = (const float*)d_in[3];
    const float* b  = (const float*)d_in[4];
    float* out = (float*)d_out;

    char* ws = (char*)d_ws;
    float*    z    = (float*)   (ws + Z_OFF);
    _Float16* z_ht = (_Float16*)(ws + ZHT_OFF);
    float*    si   = (float*)   (ws + SI_OFF);
    float*    sj   = (float*)   (ws + SJ_OFF);
    _Float16* w_ht = (_Float16*)(ws + WHT_OFF);

    prep_w   <<<64,  256, 0, stream>>>(W, w_ht);
    gemm_z   <<<NN / 16, 32, 0, stream>>>(x, w_ht, z, z_ht);
    scores   <<<NN / 256, 256, 0, stream>>>(z, a, si, sj);
    gat_fused<<<NN / 16, 128, 0, stream>>>(adj, si, sj, z_ht, b, out);
}
